// WindowAttention_34540126994811
// MI455X (gfx1250) — compile-verified, matrix operand order checked by reading, not run on hardware
//
#include <hip/hip_runtime.h>
#include <hip/hip_bf16.h>
#include <math.h>

// Problem constants (from reference)
#define DIMC    96
#define HEADS   6
#define HDIM    16
#define KNBR    16
#define QGLC    16       // WINDOW/QUANT
#define TBLN    64       // 4*QGL table rows
// SCALE = HDIM^-0.5 = 0.25, K (GEMM depth) = 96 for both GEMMs

typedef _Float16 v16h  __attribute__((ext_vector_type(16)));
typedef float    v8f   __attribute__((ext_vector_type(8)));

// ---------------------------------------------------------------------------
// xyz min reduction (xyz >= 0, so IEEE bits order as unsigned ints)
// ---------------------------------------------------------------------------
__global__ void init_mins_kernel(unsigned* mins) {
  if (threadIdx.x < 3) mins[threadIdx.x] = 0x7F800000u; // +inf
}

__global__ void xyz_min_kernel(const float* __restrict__ xyz, unsigned* mins, int n3) {
  int t = blockIdx.x * blockDim.x + threadIdx.x;
  if (t < n3) atomicMin(&mins[t % 3], __float_as_uint(xyz[t]));
}

// xq = floor(((xyz - min + shift) mod 4.0) / 0.25), per point, per axis
__global__ void quant_kernel(const float* __restrict__ xyz, const unsigned* __restrict__ mins,
                             const float* __restrict__ shift, int* __restrict__ xq, int n3) {
  int t = blockIdx.x * blockDim.x + threadIdx.x;
  if (t >= n3) return;
  float x = xyz[t] - __uint_as_float(mins[t % 3]) + shift[0];
  float r = x - floorf(x * 0.25f) * 4.0f;   // x mod 4.0 (x >= 0)
  xq[t] = (int)floorf(r * 4.0f);            // / 0.25
}

// ---------------------------------------------------------------------------
// Fragment packing.  WMMA 16-bit operand layout (A 16x32 / B 32x16):
//   lane<16 : row/col = lane,     K slots {off..off+7, 16+off..16+off+7}, off=0
//   lane>=16: row/col = lane-16,  same with off=8
// Fragment = 32 lanes x 16 halves (contiguous 32B per lane) = 1024 B.
// ---------------------------------------------------------------------------

// A[M x 96] (f32, row-major) -> fragments f = tm*3 + kc, converting to f16.
__global__ void pack_a_kernel(const float* __restrict__ src, _Float16* __restrict__ dst,
                              int rowTiles) {
  int t = blockIdx.x * blockDim.x + threadIdx.x;   // one thread = one lane-fragment
  if (t >= rowTiles * 3 * 32) return;
  int lane = t & 31;
  int f    = t >> 5;                 // tm*3 + kc
  int tm   = f / 3, kc = f - tm * 3;
  int row  = (tm << 4) + (lane & 15);
  int off8 = (lane >> 4) << 3;       // 0 or 8
  const float* s0 = src + (size_t)row * DIMC + kc * 32 + off8;  // slots 0..7 (contig)
  const float* s1 = s0 + 16;                                    // slots 8..15 (contig)
  _Float16* d = dst + ((size_t)f << 9) + lane * 16;
#pragma unroll
  for (int u = 0; u < 8; ++u) { d[u] = (_Float16)s0[u]; d[8 + u] = (_Float16)s1[u]; }
}

// B[96 x Ncols] (f32, row-major) -> fragments f = kc*tilesN + tn (tiny, one-time).
__global__ void pack_b_kernel(const float* __restrict__ src, _Float16* __restrict__ dst,
                              int Ncols) {
  int tilesN = Ncols >> 4;
  int t = blockIdx.x * blockDim.x + threadIdx.x;
  if (t >= 3 * tilesN * 32) return;
  int lane = t & 31;
  int f    = t >> 5;                 // kc*tilesN + tn
  int kc   = f / tilesN, tn = f - kc * tilesN;
  int col  = (tn << 4) + (lane & 15);
  int off8 = (lane >> 4) << 3;
  _Float16* d = dst + ((size_t)f << 9) + lane * 16;
#pragma unroll
  for (int u = 0; u < 8; ++u) {
    d[u]     = (_Float16)src[(size_t)(kc * 32 + off8 + u) * Ncols + col];
    d[8 + u] = (_Float16)src[(size_t)(kc * 32 + 16 + off8 + u) * Ncols + col];
  }
}

// Bias tables [idx][h][d][3] -> [c][idx][h][d] so d-runs are contiguous.
__global__ void table_repack_kernel(const float* __restrict__ src, float* __restrict__ dst,
                                    int n /* TBLN*96*3 */) {
  int t = blockIdx.x * blockDim.x + threadIdx.x;
  if (t >= n) return;
  int c    = t / (TBLN * DIMC);
  int rem  = t - c * (TBLN * DIMC);      // idx*96 + h*16 + d
  dst[t] = src[(size_t)rem * 3 + c];
}

// ---------------------------------------------------------------------------
// WMMA GEMM on pre-packed fragments: C[M x Ncols] = A * B + bias,
// columns < scaleCols scaled.  One wave -> 16x32 output (2 tiles, A reused).
// K = 96 hard-coded -> fully unrolled, 6 v_wmma per wave, all loads b128.
// ---------------------------------------------------------------------------
__global__ void gemm_wmma_kernel(const _Float16* __restrict__ Apk, const _Float16* __restrict__ Bpk,
                                 const float* __restrict__ bias, float* __restrict__ C,
                                 int Ncols, int totalWaves, int scaleCols, float scale) {
  int wave = (int)((blockIdx.x * (unsigned)blockDim.x + threadIdx.x) >> 5);
  if (wave >= totalWaves) return;        // wave-uniform guard: EXEC stays all-ones
  int lane   = threadIdx.x & 31;
  int tilesN = Ncols >> 4;
  int pairsN = tilesN >> 1;
  int tm  = wave / pairsN;
  int tn0 = (wave - tm * pairsN) << 1;

  v8f acc0 = {}, acc1 = {};
#pragma unroll
  for (int kc = 0; kc < 3; ++kc) {
    v16h a  = *(const v16h*)(Apk + (((size_t)(tm * 3 + kc)) << 9) + lane * 16);
    v16h b0 = *(const v16h*)(Bpk + (((size_t)(kc * tilesN + tn0)) << 9) + lane * 16);
    v16h b1 = *(const v16h*)(Bpk + (((size_t)(kc * tilesN + tn0 + 1)) << 9) + lane * 16);
    acc0 = __builtin_amdgcn_wmma_f32_16x16x32_f16(false, a, false, b0, (short)0, acc0, false, false);
    acc1 = __builtin_amdgcn_wmma_f32_16x16x32_f16(false, a, false, b1, (short)0, acc1, false, false);
  }

  // C/D layout: VGPR r -> row (tm*16 + r + 8*(lane>=16)), col = tile*16 + (lane&15)
  int rbase = (tm << 4) + ((lane >> 4) << 3);
  int col0  = (tn0 << 4) + (lane & 15);
  int col1  = col0 + 16;
  float bb0 = bias[col0], bb1 = bias[col1];
  float s0  = (col0 < scaleCols) ? scale : 1.0f;
  float s1  = (col1 < scaleCols) ? scale : 1.0f;
#pragma unroll
  for (int r = 0; r < 8; ++r) {
    C[(size_t)(rbase + r) * Ncols + col0] = (acc0[r] + bb0) * s0;
    C[(size_t)(rbase + r) * Ncols + col1] = (acc1[r] + bb1) * s1;
  }
}

// ---------------------------------------------------------------------------
// Attention: one thread per (point, head), single pass, online softmax.
// All gathered runs (k row, v row, repacked tables) are 64B-contiguous.
// ---------------------------------------------------------------------------
__global__ void attn_kernel(const float* __restrict__ qkv, const int* __restrict__ index_1,
                            const int* __restrict__ xq,
                            const float* __restrict__ tqT, const float* __restrict__ tkT,
                            const float* __restrict__ tvT,
                            float* __restrict__ xout, int N) {
  int t = blockIdx.x * blockDim.x + threadIdx.x;
  if (t >= N * HEADS) return;
  int i = t / HEADS, h = t - i * HEADS;

  float q[HDIM];
  const float* qrow = qkv + (size_t)i * (3 * DIMC) + h * HDIM;  // q already *SCALE
#pragma unroll
  for (int d = 0; d < HDIM; ++d) q[d] = qrow[d];
  int xi0 = xq[i * 3 + 0], xi1 = xq[i * 3 + 1], xi2 = xq[i * 3 + 2];

  float mr = -3.0e38f, denom = 0.0f;
  float out[HDIM];
#pragma unroll
  for (int d = 0; d < HDIM; ++d) out[d] = 0.0f;

  for (int jj = 0; jj < KNBR; ++jj) {
    int j = index_1[i * KNBR + jj];
    const float* krow = qkv + (size_t)j * (3 * DIMC) + DIMC + h * HDIM;
    const float* vrow = krow + DIMC;
    int ri[3];
    ri[0] = xi0 - xq[j * 3 + 0] + (QGLC - 1);
    ri[1] = xi1 - xq[j * 3 + 1] + (QGLC - 1);
    ri[2] = xi2 - xq[j * 3 + 2] + (QGLC - 1);

    float kv[HDIM], vv[HDIM];
    float s = 0.0f;
#pragma unroll
    for (int d = 0; d < HDIM; ++d) { kv[d] = krow[d]; vv[d] = vrow[d]; s += q[d] * kv[d]; }
#pragma unroll
    for (int c = 0; c < 3; ++c) {
      size_t base = ((size_t)(c * TBLN + ri[c])) * DIMC + h * HDIM;  // contiguous in d
      const float* tqp = tqT + base;
      const float* tkp = tkT + base;
      const float* tvp = tvT + base;
#pragma unroll
      for (int d = 0; d < HDIM; ++d) {
        s     += q[d] * tqp[d] + kv[d] * tkp[d];
        vv[d] += tvp[d];
      }
    }
    // online softmax update
    float mnew = fmaxf(mr, s);
    float corr = __expf(mr - mnew);
    float w    = __expf(s - mnew);
    denom = denom * corr + w;
#pragma unroll
    for (int d = 0; d < HDIM; ++d) out[d] = out[d] * corr + w * vv[d];
    mr = mnew;
  }

  float inv = 1.0f / denom;
  float* xo = xout + (size_t)i * DIMC + h * HDIM;
#pragma unroll
  for (int d = 0; d < HDIM; ++d) xo[d] = out[d] * inv;
}

// ---------------------------------------------------------------------------
extern "C" void kernel_launch(void* const* d_in, const int* in_sizes, int n_in,
                              void* d_out, int out_size, void* d_ws, size_t ws_size,
                              hipStream_t stream) {
  const float* feats   = (const float*)d_in[0];
  const float* xyz     = (const float*)d_in[1];
  // d_in[2] index_0, d_in[3] index_0_offsets, d_in[4] n_max: implied by fixed K=16 layout
  const int*   index_1 = (const int*)d_in[5];
  const float* shift   = (const float*)d_in[6];
  const float* W_qkv   = (const float*)d_in[7];
  const float* b_qkv   = (const float*)d_in[8];
  const float* table_q = (const float*)d_in[9];
  const float* table_k = (const float*)d_in[10];
  const float* table_v = (const float*)d_in[11];
  const float* W_proj  = (const float*)d_in[12];
  const float* b_proj  = (const float*)d_in[13];
  (void)n_in; (void)out_size; (void)ws_size;

  const int N        = in_sizes[0] / DIMC;  // 50000 (divisible by 16)
  const int n3       = N * 3;
  const int rowTiles = N / 16;
  const int tblElems = TBLN * DIMC * 3;     // 18432 per table

  // Workspace carve-out
  char* ws = (char*)d_ws;
  size_t off = 0;
  auto take = [&](size_t bytes) -> char* {
    char* p = ws + off;
    off = (off + bytes + 255) & ~(size_t)255;
    return p;
  };
  unsigned*  mins    = (unsigned*)take(16);
  int*       xq      = (int*)take((size_t)n3 * sizeof(int));
  _Float16*  aPk     = (_Float16*)take((size_t)rowTiles * 3 * 1024);        // packed feats
  _Float16*  xPk     = (_Float16*)take((size_t)rowTiles * 3 * 1024);        // packed attn out
  _Float16*  wqkvPk  = (_Float16*)take((size_t)3 * 18 * 1024);
  _Float16*  wprojPk = (_Float16*)take((size_t)3 * 6 * 1024);
  float*     tqT     = (float*)take((size_t)tblElems * sizeof(float));
  float*     tkT     = (float*)take((size_t)tblElems * sizeof(float));
  float*     tvT     = (float*)take((size_t)tblElems * sizeof(float));
  float*     qkv     = (float*)take((size_t)N * 3 * DIMC * sizeof(float));
  float*     xF      = (float*)take((size_t)N * DIMC * sizeof(float));

  // 1) xyz mins + quantized coords
  hipLaunchKernelGGL(init_mins_kernel, dim3(1), dim3(32), 0, stream, mins);
  hipLaunchKernelGGL(xyz_min_kernel, dim3((n3 + 255) / 256), dim3(256), 0, stream, xyz, mins, n3);
  hipLaunchKernelGGL(quant_kernel, dim3((n3 + 255) / 256), dim3(256), 0, stream,
                     xyz, mins, shift, xq, n3);

  // 2) repack bias tables (tiny)
  hipLaunchKernelGGL(table_repack_kernel, dim3((tblElems + 255) / 256), dim3(256), 0, stream,
                     table_q, tqT, tblElems);
  hipLaunchKernelGGL(table_repack_kernel, dim3((tblElems + 255) / 256), dim3(256), 0, stream,
                     table_k, tkT, tblElems);
  hipLaunchKernelGGL(table_repack_kernel, dim3((tblElems + 255) / 256), dim3(256), 0, stream,
                     table_v, tvT, tblElems);

  // 3) pack GEMM operands into WMMA fragment layout (converts f32 -> f16)
  {
    int tA = rowTiles * 3 * 32;
    hipLaunchKernelGGL(pack_a_kernel, dim3((tA + 255) / 256), dim3(256), 0, stream,
                       feats, aPk, rowTiles);
    int tBq = 3 * 18 * 32;
    hipLaunchKernelGGL(pack_b_kernel, dim3((tBq + 255) / 256), dim3(256), 0, stream,
                       W_qkv, wqkvPk, 3 * DIMC);
    int tBp = 3 * 6 * 32;
    hipLaunchKernelGGL(pack_b_kernel, dim3((tBp + 255) / 256), dim3(256), 0, stream,
                       W_proj, wprojPk, DIMC);
  }

  // 4) QKV projection: (N x 96) @ (96 x 288) + b, q-columns (<96) scaled by 0.25
  {
    int waves = rowTiles * (18 / 2);
    hipLaunchKernelGGL(gemm_wmma_kernel, dim3((waves + 3) / 4), dim3(128), 0, stream,
                       aPk, wqkvPk, b_qkv, qkv, 3 * DIMC, waves, DIMC, 0.25f);
  }

  // 5) Gather + bias + online-softmax + V aggregation
  {
    int threads = N * HEADS;
    hipLaunchKernelGGL(attn_kernel, dim3((threads + 255) / 256), dim3(256), 0, stream,
                       qkv, index_1, xq, tqT, tkT, tvT, xF, N);
  }

  // 6) pack attention output, then proj GEMM -> d_out (f32)
  {
    int tA = rowTiles * 3 * 32;
    hipLaunchKernelGGL(pack_a_kernel, dim3((tA + 255) / 256), dim3(256), 0, stream,
                       xF, xPk, rowTiles);
    int waves = rowTiles * (6 / 2);
    hipLaunchKernelGGL(gemm_wmma_kernel, dim3((waves + 3) / 4), dim3(128), 0, stream,
                       xPk, wprojPk, b_proj, (float*)d_out, DIMC, waves, 0, 1.0f);
  }
}
